// SSCDModule_26654567039550
// MI455X (gfx1250) — matrix operand order repaired and checked by reading, not compiled, not run on hardware
//
#include <hip/hip_runtime.h>

// ---------------------------------------------------------------------------
// MI455X (gfx1250) hypercorrelation encoder.
// All convolutions (CPConv4d branch1/branch2 + decoder) are implicit GEMMs on
// the WMMA pipe (v_wmma_f32_16x16x32_f16, f32 accumulate), with weights staged
// into LDS (optionally via the Tensor Data Mover) as pre-transposed f16.
// ---------------------------------------------------------------------------

typedef __attribute__((ext_vector_type(16))) _Float16 v16h;
typedef __attribute__((ext_vector_type(8)))  float    v8f;
typedef __attribute__((ext_vector_type(4)))  unsigned int v4u;
typedef __attribute__((ext_vector_type(8)))  int      v8i_;
typedef __attribute__((ext_vector_type(4)))  int      v4i_;

#define WAVES_PER_BLOCK 8
#define CONV_BLOCK (WAVES_PER_BLOCK * 32)
#define GN_NGROUPS 4

__device__ __forceinline__ int imin_d(int a, int b) { return a < b ? a : b; }

// Generic batched implicit-GEMM convolution over a strided 6D view.
//   input element:  b*iB + s1*iS1 + s0*iS0 + ic*iC + y*iY + x*iX
//   output element: b*oB + s1*oS1 + s0*oS0 + oc*oC + y*oY + x*oX
// M = global output pixel over (b, s1, s0, y, x); N = oc; K = IC*k*k.
struct ConvParams {
    const float* in;
    const float* w;     // [OC, IC, KK, KK]
    const float* bias;  // [OC]
    float* out;
    long long iB, iS1, iS0, iC, iY, iX;
    long long oB, oS1, oS0, oC, oY, oX;
    int IH, IW, IC, KK, PAD, STR;
    int OH, OW, OC;
    int Sb, Ws;         // images per batch = Sb, s1 = s/Ws, s0 = s%Ws
    int Mtotal, Ktot;
    int accumulate;     // 1: read existing out into C (branch2 adds onto branch1)
    int relu;
    // fast-path extras
    int icLog, Kpad, OCpad, useTdm, rawElems;
    unsigned f16Off;    // byte offset of f16 weight region in dynamic LDS
};

__device__ __forceinline__ long long conv_oaddr(const ConvParams& p, int m, int oc) {
    int img = m / (p.OH * p.OW);
    int pix = m - img * (p.OH * p.OW);
    int b = img / p.Sb, s = img - b * p.Sb;
    int s1 = s / p.Ws, s0 = s - s1 * p.Ws;
    int y = pix / p.OW, x = pix - y * p.OW;
    return (long long)b * p.oB + (long long)s1 * p.oS1 + (long long)s0 * p.oS0 +
           (long long)oc * p.oC + (long long)y * p.oY + (long long)x * p.oX;
}

// --------------------------- generic (non-pow2 IC) --------------------------
__global__ void __launch_bounds__(CONV_BLOCK)
conv_wmma_kernel(ConvParams p) {
    const int lane = threadIdx.x & 31;
    const int wave = threadIdx.x >> 5;
    const int nMt = (p.Mtotal + 15) >> 4;
    const int nNt = (p.OC + 15) >> 4;
    const int tile = blockIdx.x * WAVES_PER_BLOCK + wave;   // wave-uniform
    if (tile >= nMt * nNt) return;
    const int tileM = tile % nMt;
    const int tileN = tile / nMt;
    const int half = lane >> 4;
    const int l16  = lane & 15;
    const int R = p.KK * p.KK;

    const int mA = tileM * 16 + l16;
    const bool mAv = (mA < p.Mtotal);
    long long inbase = 0;
    int y0 = 0, x0 = 0;
    if (mAv) {
        int img = mA / (p.OH * p.OW);
        int pix = mA - img * (p.OH * p.OW);
        int b = img / p.Sb, s = img - b * p.Sb;
        int s1 = s / p.Ws, s0 = s - s1 * p.Ws;
        inbase = (long long)b * p.iB + (long long)s1 * p.iS1 + (long long)s0 * p.iS0;
        int oy = pix / p.OW, ox = pix - oy * p.OW;
        y0 = oy * p.STR - p.PAD;
        x0 = ox * p.STR - p.PAD;
    }
    const int oc = tileN * 16 + l16;
    const bool ocv = (oc < p.OC);
    const long long wbase = (long long)oc * p.IC * R;

    v8f c;
    const float bs = ocv ? p.bias[oc] : 0.f;
#pragma unroll
    for (int r = 0; r < 8; ++r) c[r] = bs;
    if (p.accumulate) {
#pragma unroll
        for (int r = 0; r < 8; ++r) {
            int m = tileM * 16 + (half << 3) + r;
            if (m < p.Mtotal && ocv) c[r] += p.out[conv_oaddr(p, m, oc)];
        }
    }

    for (int kc = 0; kc < p.Ktot; kc += 32) {
        v16h a, bf;
#pragma unroll
        for (int e = 0; e < 16; ++e) {
            const int v = e >> 1;
            int kA = kc + ((v & 3) << 1) + ((v & 4) ? 16 : 0) + (half << 3) + (e & 1);
            float av = 0.f;
            if (mAv && kA < p.Ktot) {
                int ic = kA / R, t = kA - ic * R;
                int ky = t / p.KK, kx = t - ky * p.KK;
                int iy = y0 + ky, ix = x0 + kx;
                if (iy >= 0 && iy < p.IH && ix >= 0 && ix < p.IW)
                    av = p.in[inbase + (long long)ic * p.iC +
                              (long long)iy * p.iY + (long long)ix * p.iX];
            }
            a[e] = (_Float16)av;
            int kB = kc + ((v & 3) << 1) + ((v & 4) ? 16 : 0) + (half << 3) + (e & 1);
            float bv = (ocv && kB < p.Ktot) ? p.w[wbase + kB] : 0.f;
            bf[e] = (_Float16)bv;
        }
        c = __builtin_amdgcn_wmma_f32_16x16x32_f16(false, a, false, bf,
                                                   (short)0, c, false, false);
    }
#pragma unroll
    for (int r = 0; r < 8; ++r) {
        int m = tileM * 16 + (half << 3) + r;
        if (m < p.Mtotal && ocv) {
            float v = c[r];
            if (p.relu) v = fmaxf(v, 0.f);
            p.out[conv_oaddr(p, m, oc)] = v;
        }
    }
}

// ---------------------- fast path (IC power-of-two) -------------------------
// K ordering: k = tap*IC + ic.  LDS weights: w16[oc*Kpad + k] (f16, zero-padded).
// Optional: raw f32 weights DMA'd to LDS via the Tensor Data Mover first.
__device__ __forceinline__ void tdm_load_weights(const float* w, int nelems) {
    unsigned long long ga = (unsigned long long)(uintptr_t)w;
    v4u g0;
    g0.x = 1u;                                    // count=1 (valid descriptor)
    g0.y = 0u;                                    // lds_addr = 0 (start of dyn-LDS)
    g0.z = (unsigned)(ga & 0xffffffffu);          // global_addr[31:0]
    g0.w = (unsigned)((ga >> 32) & 0x01ffffffu) | (2u << 30);  // addr[56:32] | type=2
    unsigned n = (unsigned)nelems;
    v8i_ g1 = {0, 0, 0, 0, 0, 0, 0, 0};
    g1[0] = (2 << 16);                            // data_size = 4 bytes
    g1[1] = (int)((n & 0xffffu) << 16);           // tensor_dim0[15:0] @ bits 48..63
    g1[2] = (int)(((n >> 16) & 0xffffu) | (1u << 16)); // dim0[31:16] | tensor_dim1=1
    g1[3] = (int)((n & 0xffffu) << 16);           // tile_dim0 = n  @ bits 112..127
    g1[4] = 1;                                    // tile_dim1 = 1, tile_dim2 = 0
    g1[5] = (int)n;                               // tensor_dim0_stride[31:0]
    v4i_ g2 = {0, 0, 0, 0};
    v4i_ g3 = {0, 0, 0, 0};
#if defined(__clang_major__) && (__clang_major__ >= 23)
    v8i_ g4 = {0, 0, 0, 0, 0, 0, 0, 0};
    __builtin_amdgcn_tensor_load_to_lds(g0, g1, g2, g3, g4, 0);
#else
    __builtin_amdgcn_tensor_load_to_lds(g0, g1, g2, g3, 0);
#endif
    __builtin_amdgcn_s_wait_tensorcnt(0);
}

template <int KK, int NB>
__global__ void __launch_bounds__(CONV_BLOCK)
conv_wmma_fast(ConvParams p) {
    extern __shared__ unsigned char smem[];
    float* wraw = (float*)smem;                       // raw f32 weights (if useTdm)
    _Float16* w16 = (_Float16*)(smem + p.f16Off);     // [OCpad][Kpad] f16

    const int tid = threadIdx.x;
    const int lane = tid & 31;
    const int wave = tid >> 5;
    const int R = KK * KK;
    const int nMt = (p.Mtotal + 15) >> 4;
    const int nNt = (p.OC + 16 * NB - 1) / (16 * NB);
    const int tile = blockIdx.x * WAVES_PER_BLOCK + wave;
    const bool active = tile < nMt * nNt;

    // ---- stage weights into LDS (all waves participate; barriers uniform) ----
    if (p.useTdm) {
        if (tid == 0) tdm_load_weights(p.w, p.rawElems);  // TDM -> LDS[0..)
        __syncthreads();
    }
    {
        const int ICR = p.IC * R;
        const int icMask = (1 << p.icLog) - 1;
        const int tot = p.OCpad * p.Kpad;
        for (int idx = tid; idx < tot; idx += CONV_BLOCK) {
            int oc = idx / p.Kpad, k = idx - oc * p.Kpad;
            float v = 0.f;
            if (oc < p.OC && k < p.Ktot) {
                int ic = k & icMask;
                int tap = k >> p.icLog;
                int si = oc * ICR + ic * R + tap;
                v = p.useTdm ? wraw[si] : p.w[si];
            }
            w16[idx] = (_Float16)v;
        }
    }
    __syncthreads();
    if (!active) return;

    const int tileM = tile % nMt;
    const int tileN = tile / nMt;
    const int half = lane >> 4;
    const int l16  = lane & 15;

    const int mA = tileM * 16 + l16;
    const bool mAv = (mA < p.Mtotal);
    long long inbase = 0;
    int y0 = 0, x0 = 0;
    if (mAv) {
        int img = mA / (p.OH * p.OW);
        int pix = mA - img * (p.OH * p.OW);
        int b = img / p.Sb, s = img - b * p.Sb;
        int s1 = s / p.Ws, s0 = s - s1 * p.Ws;
        inbase = (long long)b * p.iB + (long long)s1 * p.iS1 + (long long)s0 * p.iS0;
        int oy = pix / p.OW, ox = pix - oy * p.OW;
        y0 = oy * p.STR - p.PAD;
        x0 = ox * p.STR - p.PAD;
    }

    const int ocb = tileN * (16 * NB);
    v8f c[NB];
#pragma unroll
    for (int nb = 0; nb < NB; ++nb) {
        int oc = ocb + nb * 16 + l16;
        float bs = (oc < p.OC) ? p.bias[oc] : 0.f;
#pragma unroll
        for (int r = 0; r < 8; ++r) c[nb][r] = bs;
        if (p.accumulate) {
#pragma unroll
            for (int r = 0; r < 8; ++r) {
                int m = tileM * 16 + (half << 3) + r;
                if (m < p.Mtotal && oc < p.OC) c[nb][r] += p.out[conv_oaddr(p, m, oc)];
            }
        }
    }

    const int icMask = (1 << p.icLog) - 1;
    for (int kc = 0; kc < p.Ktot; kc += 32) {
        // A fragment: 16 gathered f32 -> f16 (k = tap*IC + ic; KK is constexpr)
        v16h a;
#pragma unroll
        for (int e = 0; e < 16; ++e) {
            const int v = e >> 1;
            int kA = kc + ((v & 3) << 1) + ((v & 4) ? 16 : 0) + (half << 3) + (e & 1);
            float av = 0.f;
            if (mAv && kA < p.Ktot) {
                int ic = kA & icMask;
                int tap = kA >> p.icLog;
                int ky = tap / KK;            // constant divisor -> mul/shift
                int kx = tap - ky * KK;
                int iy = y0 + ky, ix = x0 + kx;
                if (iy >= 0 && iy < p.IH && ix >= 0 && ix < p.IW)
                    av = p.in[inbase + (long long)ic * p.iC +
                              (long long)iy * p.iY + (long long)ix * p.iX];
            }
            a[e] = (_Float16)av;
        }
        // NB matrix ops sharing one A fragment; B from LDS as packed b32 pairs
#pragma unroll
        for (int nb = 0; nb < NB; ++nb) {
            const int oc = ocb + nb * 16 + l16;              // < OCpad by construction
            const unsigned* wrow = (const unsigned*)(w16 + (size_t)oc * p.Kpad);
            v8i_ bi;
#pragma unroll
            for (int v2 = 0; v2 < 8; ++v2) {
                int kB = kc + ((v2 & 3) << 1) + ((v2 & 4) ? 16 : 0) + (half << 3);       // even
                bi[v2] = (int)wrow[kB >> 1];
            }
            v16h bf = __builtin_bit_cast(v16h, bi);
            c[nb] = __builtin_amdgcn_wmma_f32_16x16x32_f16(false, a, false, bf,
                                                           (short)0, c[nb], false, false);
        }
    }

#pragma unroll
    for (int nb = 0; nb < NB; ++nb) {
        int oc = ocb + nb * 16 + l16;
#pragma unroll
        for (int r = 0; r < 8; ++r) {
            int m = tileM * 16 + (half << 3) + r;
            if (m < p.Mtotal && oc < p.OC) {
                float v = c[nb][r];
                if (p.relu) v = fmaxf(v, 0.f);
                p.out[conv_oaddr(p, m, oc)] = v;
            }
        }
    }
}

// ---------------------------------------------------------------------------
// GroupNorm (4 groups) + ReLU, two-phase with f32 atomics.
// ---------------------------------------------------------------------------
__global__ void zero_kernel(float* p, int n) {
    int i = blockIdx.x * blockDim.x + threadIdx.x;
    if (i < n) p[i] = 0.f;
}

__global__ void gn_partial_kernel(const float* __restrict__ x, float* __restrict__ sums,
                                  int C, long long rest, long long total, int npairs) {
    float ls[8], ls2[8];
#pragma unroll
    for (int i = 0; i < 8; ++i) { ls[i] = 0.f; ls2[i] = 0.f; }
    const int cg = C / GN_NGROUPS;
    const long long stride = (long long)gridDim.x * blockDim.x;
    for (long long i = (long long)blockIdx.x * blockDim.x + threadIdx.x; i < total; i += stride) {
        int c = (int)((i / rest) % C);
        int b = (int)(i / (rest * C));
        int pr = b * GN_NGROUPS + c / cg;
        float v = x[i];
        ls[pr] += v;
        ls2[pr] += v * v;
    }
    __shared__ float ss[16];
    if (threadIdx.x < 16) ss[threadIdx.x] = 0.f;
    __syncthreads();
    for (int i = 0; i < npairs; ++i) {
        atomicAdd(&ss[2 * i], ls[i]);
        atomicAdd(&ss[2 * i + 1], ls2[i]);
    }
    __syncthreads();
    if ((int)threadIdx.x < 2 * npairs) atomicAdd(&sums[threadIdx.x], ss[threadIdx.x]);
}

__global__ void gn_apply_kernel(float* __restrict__ x, const float* __restrict__ sums,
                                const float* __restrict__ gamma, const float* __restrict__ beta,
                                int C, long long rest, long long total, float eps) {
    long long i = (long long)blockIdx.x * blockDim.x + threadIdx.x;
    if (i >= total) return;
    int c = (int)((i / rest) % C);
    int b = (int)(i / (rest * C));
    int cg = C / GN_NGROUPS;
    int pr = b * GN_NGROUPS + c / cg;
    float n = (float)((long long)cg * rest);
    float mean = sums[2 * pr] / n;
    float var = sums[2 * pr + 1] / n - mean * mean;
    float v = (x[i] - mean) * rsqrtf(var + eps);
    v = v * gamma[c] + beta[c];
    x[i] = fmaxf(v, 0.f);   // GN is always followed by ReLU here
}

// ---------------------------------------------------------------------------
// Bilinear (align_corners=True) over the query dims of [B,C,H,W,Hb,Wb],
// optionally fused with an elementwise addend. Hb=Wb=1 covers the 2D case.
// ---------------------------------------------------------------------------
__global__ void interp6d_kernel(const float* __restrict__ in, const float* __restrict__ add,
                                float* __restrict__ out,
                                int B, int C, int H, int W, int Hb, int Wb,
                                int OH, int OW, long long total) {
    long long i = (long long)blockIdx.x * blockDim.x + threadIdx.x;
    if (i >= total) return;
    long long t = i;
    int wb = (int)(t % Wb); t /= Wb;
    int hb = (int)(t % Hb); t /= Hb;
    int xa = (int)(t % OW); t /= OW;
    int ya = (int)(t % OH); t /= OH;
    int c  = (int)(t % C);  t /= C;
    int b  = (int)t;
    float ys = (OH > 1 && H > 1) ? (float)ya * (float)(H - 1) / (float)(OH - 1) : 0.f;
    float xs = (OW > 1 && W > 1) ? (float)xa * (float)(W - 1) / (float)(OW - 1) : 0.f;
    int y0 = (int)floorf(ys); int y1 = imin_d(y0 + 1, H - 1); float wy = ys - (float)y0;
    int x0 = (int)floorf(xs); int x1 = imin_d(x0 + 1, W - 1); float wx = xs - (float)x0;
    long long bc = (long long)b * C + c;
    auto at = [&](int y, int x) -> float {
        return in[(((bc * H + y) * W + x) * Hb + hb) * Wb + wb];
    };
    float v = (1.f - wy) * ((1.f - wx) * at(y0, x0) + wx * at(y0, x1)) +
                     wy  * ((1.f - wx) * at(y1, x0) + wx * at(y1, x1));
    if (add) v += add[i];
    out[i] = v;
}

// mean over support dims (size S) of [B,64,HaWa,S], duplicated to 128 channels.
__global__ void mean_concat_kernel(const float* __restrict__ x, float* __restrict__ out,
                                   int HaWa, int S, long long total) {
    long long i = (long long)blockIdx.x * blockDim.x + threadIdx.x;
    if (i >= total) return;
    long long t = i;
    int p = (int)(t % HaWa); t /= HaWa;
    int c = (int)(t % 128);  t /= 128;
    int b = (int)t;
    int cc = c & 63;
    long long base = (((long long)b * 64 + cc) * HaWa + p) * S;
    float s = 0.f;
    for (int k = 0; k < S; ++k) s += x[base + k];
    out[i] = s / (float)S;
}

// ---------------------------------------------------------------------------
// Host-side orchestration
// ---------------------------------------------------------------------------
struct Shape6 { int B, C, Ha, Wa, Hb, Wb; };

static void launch_conv_any(hipStream_t st, ConvParams p) {
    const bool pow2 = (p.IC & (p.IC - 1)) == 0 && p.IC > 0 && (p.KK == 3 || p.KK == 5);
    if (!pow2) {
        long long tiles = (long long)((p.Mtotal + 15) / 16) * ((p.OC + 15) / 16);
        int blocks = (int)((tiles + WAVES_PER_BLOCK - 1) / WAVES_PER_BLOCK);
        conv_wmma_kernel<<<blocks, CONV_BLOCK, 0, st>>>(p);
        return;
    }
    int NB = 1;
    if (p.OC >= 64) NB = 4;
    else if (p.OC >= 32) NB = 2;
    int icLog = 0;
    while ((1 << icLog) < p.IC) ++icLog;
    p.icLog = icLog;
    p.Kpad = (p.Ktot + 31) & ~31;
    p.OCpad = ((p.OC + 16 * NB - 1) / (16 * NB)) * 16 * NB;
    p.rawElems = p.OC * p.IC * p.KK * p.KK;
    size_t rawB = (size_t)p.rawElems * 4;
    size_t f16B = (size_t)p.OCpad * p.Kpad * 2;
    p.useTdm = (rawB + f16B) <= (size_t)(224 * 1024) ? 1 : 0;
    p.f16Off = p.useTdm ? (unsigned)rawB : 0u;
    size_t lds = (size_t)p.f16Off + f16B;
    long long tiles = (long long)((p.Mtotal + 15) / 16) *
                      ((p.OC + 16 * NB - 1) / (16 * NB));
    int blocks = (int)((tiles + WAVES_PER_BLOCK - 1) / WAVES_PER_BLOCK);
    if (p.KK == 3) {
        if (NB == 4)      conv_wmma_fast<3, 4><<<blocks, CONV_BLOCK, lds, st>>>(p);
        else if (NB == 2) conv_wmma_fast<3, 2><<<blocks, CONV_BLOCK, lds, st>>>(p);
        else              conv_wmma_fast<3, 1><<<blocks, CONV_BLOCK, lds, st>>>(p);
    } else {
        if (NB == 4)      conv_wmma_fast<5, 4><<<blocks, CONV_BLOCK, lds, st>>>(p);
        else if (NB == 2) conv_wmma_fast<5, 2><<<blocks, CONV_BLOCK, lds, st>>>(p);
        else              conv_wmma_fast<5, 1><<<blocks, CONV_BLOCK, lds, st>>>(p);
    }
}

static void cpconv_layer(hipStream_t st, const float* in, Shape6 s, float* out,
                         int oc, int k, int stride,
                         const float* w1, const float* b1,
                         const float* w2, const float* b2,
                         const float* gg, const float* gb,
                         float* sums, Shape6* outShape) {
    const int pad = k / 2;
    const int HbO = (s.Hb + stride - 1) / stride;
    const int WbO = (s.Wb + stride - 1) / stride;
    const long long iC = (long long)s.Ha * s.Wa * s.Hb * s.Wb;
    const long long iB = (long long)s.C * iC;
    const long long oC = (long long)s.Ha * s.Wa * HbO * WbO;
    const long long oB = (long long)oc * oC;

    // branch1: conv over query dims (Ha,Wa); images = (b, yb_out, xb_out)
    ConvParams p1{};
    p1.in = in; p1.w = w1; p1.bias = b1; p1.out = out;
    p1.iB = iB; p1.iS1 = (long long)stride * s.Wb; p1.iS0 = stride;
    p1.iC = iC; p1.iY = (long long)s.Wa * s.Hb * s.Wb; p1.iX = (long long)s.Hb * s.Wb;
    p1.oB = oB; p1.oC = oC;
    p1.oY = (long long)s.Wa * HbO * WbO; p1.oX = (long long)HbO * WbO;
    p1.oS1 = WbO; p1.oS0 = 1;
    p1.IH = s.Ha; p1.IW = s.Wa; p1.IC = s.C; p1.KK = k; p1.PAD = pad; p1.STR = 1;
    p1.OH = s.Ha; p1.OW = s.Wa; p1.OC = oc;
    p1.Sb = HbO * WbO; p1.Ws = WbO;
    p1.Mtotal = s.B * p1.Sb * s.Ha * s.Wa;
    p1.Ktot = s.C * k * k;
    p1.accumulate = 0; p1.relu = 0;
    launch_conv_any(st, p1);

    // branch2: strided conv over support dims (Hb,Wb); images = (b, ya, xa)
    ConvParams p2{};
    p2.in = in; p2.w = w2; p2.bias = b2; p2.out = out;
    p2.iB = iB; p2.iS1 = (long long)s.Wa * s.Hb * s.Wb; p2.iS0 = (long long)s.Hb * s.Wb;
    p2.iC = iC; p2.iY = s.Wb; p2.iX = 1;
    p2.oB = oB; p2.oC = oC;
    p2.oS1 = (long long)s.Wa * HbO * WbO; p2.oS0 = (long long)HbO * WbO;
    p2.oY = WbO; p2.oX = 1;
    p2.IH = s.Hb; p2.IW = s.Wb; p2.IC = s.C; p2.KK = k; p2.PAD = pad; p2.STR = stride;
    p2.OH = HbO; p2.OW = WbO; p2.OC = oc;
    p2.Sb = s.Ha * s.Wa; p2.Ws = s.Wa;
    p2.Mtotal = s.B * p2.Sb * HbO * WbO;
    p2.Ktot = s.C * k * k;
    p2.accumulate = 1; p2.relu = 0;
    launch_conv_any(st, p2);

    // GroupNorm + ReLU (in place)
    Shape6 os{ s.B, oc, s.Ha, s.Wa, HbO, WbO };
    long long rest = (long long)os.Ha * os.Wa * os.Hb * os.Wb;
    long long total = (long long)os.B * oc * rest;
    zero_kernel<<<1, 32, 0, st>>>(sums, 16);
    long long want = (total + 255) / 256;
    int pb = (int)(want < 1024 ? want : 1024);
    if (pb < 1) pb = 1;
    gn_partial_kernel<<<pb, 256, 0, st>>>(out, sums, oc, rest, total, os.B * GN_NGROUPS);
    gn_apply_kernel<<<(unsigned)((total + 255) / 256), 256, 0, st>>>(
        out, sums, gg, gb, oc, rest, total, 1e-5f);
    *outShape = os;
}

struct LayerP { const float *w1, *b1, *w2, *b2, *gg, *gb; };
// JAX pytree (alphabetical) layer order: b1, b2, gn_b, gn_g, w1, w2
static LayerP layerP(void* const* d_in, int base, int layer) {
    int o = base + layer * 6;
    LayerP lp;
    lp.b1 = (const float*)d_in[o + 0];
    lp.b2 = (const float*)d_in[o + 1];
    lp.gb = (const float*)d_in[o + 2];
    lp.gg = (const float*)d_in[o + 3];
    lp.w1 = (const float*)d_in[o + 4];
    lp.w2 = (const float*)d_in[o + 5];
    return lp;
}

static Shape6 run_block(hipStream_t st, const float* in, Shape6 s,
                        float* o0, float* o1, float* o2,
                        void* const* d_in, int base,
                        const int* outs, const int* ks, const int* strides,
                        float* sums) {
    float* obuf[3] = { o0, o1, o2 };
    const float* cur = in;
    Shape6 cs = s;
    for (int i = 0; i < 3; ++i) {
        LayerP lp = layerP(d_in, base, i);
        Shape6 ns;
        cpconv_layer(st, cur, cs, obuf[i], outs[i], ks[i], strides[i],
                     lp.w1, lp.b1, lp.w2, lp.b2, lp.gg, lp.gb, sums, &ns);
        cur = obuf[i];
        cs = ns;
    }
    return cs;
}

static void conv2d_dec(hipStream_t st, const float* in, int B, int C, int H, int W,
                       float* out, int oc, const float* w, const float* b, int relu) {
    ConvParams p{};
    p.in = in; p.w = w; p.bias = b; p.out = out;
    p.iB = (long long)C * H * W; p.iS1 = 0; p.iS0 = 0;
    p.iC = (long long)H * W; p.iY = W; p.iX = 1;
    p.oB = (long long)oc * H * W; p.oS1 = 0; p.oS0 = 0;
    p.oC = (long long)H * W; p.oY = W; p.oX = 1;
    p.IH = H; p.IW = W; p.IC = C; p.KK = 3; p.PAD = 1; p.STR = 1;
    p.OH = H; p.OW = W; p.OC = oc;
    p.Sb = 1; p.Ws = 1;
    p.Mtotal = B * H * W; p.Ktot = C * 9;
    p.accumulate = 0; p.relu = relu;
    launch_conv_any(st, p);
}

static void launch_interp(hipStream_t st, const float* in, const float* add, float* out,
                          int B, int C, int H, int W, int Hb, int Wb, int OH, int OW) {
    long long total = (long long)B * C * OH * OW * Hb * Wb;
    interp6d_kernel<<<(unsigned)((total + 255) / 256), 256, 0, st>>>(
        in, add, out, B, C, H, W, Hb, Wb, OH, OW, total);
}

extern "C" void kernel_launch(void* const* d_in, const int* in_sizes, int n_in,
                              void* d_out, int out_size, void* d_ws, size_t ws_size,
                              hipStream_t stream) {
    (void)out_size; (void)ws_size;
    hipStream_t st = stream;

    // Locate correlation tensors among the first three inputs by element count.
    const float* corr4 = nullptr;
    const float* corr3 = nullptr;
    const float* corr2 = nullptr;
    for (int i = 0; i < 3 && i < n_in; ++i) {
        long long sz = in_sizes[i];
        if (sz == 2LL * 3 * 13 * 13 * 13 * 13)      corr4 = (const float*)d_in[i];
        else if (sz == 2LL * 6 * 25 * 25 * 25 * 25) corr3 = (const float*)d_in[i];
        else if (sz == 2LL * 4 * 50 * 50 * 50 * 50) corr2 = (const float*)d_in[i];
    }
    // params pytree bases (alphabetical): dec=3(+8), enc2=11, enc3=29, enc32=47, enc4=65, enc43=83
    const int BASE_DEC = 3, BASE_E2 = 11, BASE_E3 = 29, BASE_E32 = 47, BASE_E4 = 65, BASE_E43 = 83;

    // Workspace layout (floats)
    float* W   = (float*)d_ws;
    float* A   = W;                 // up to 13,520,000 (enc2 L1 out)
    float* Bb  = W + 13600000;      // up to  2,560,000 (enc2 L2 out)
    float* S4  = W + 16300000;      // 86,528
    float* S3  = W + 16400000;      // 320,000
    float* S2  = W + 16800000;      // 1,280,000
    float* SUM = W + 18200000;      // 16

    static const int eouts_small[3] = { 16, 32, 64 };
    static const int eouts_64[3]    = { 64, 64, 64 };
    static const int ks4[3]  = { 3, 3, 3 }, st4[3]  = { 2, 2, 2 };
    static const int ks3[3]  = { 5, 3, 3 }, st3[3]  = { 4, 2, 2 };
    static const int ks2[3]  = { 5, 5, 3 }, st2[3]  = { 4, 4, 2 };
    static const int ks1[3]  = { 3, 3, 3 }, st1[3]  = { 1, 1, 1 };

    // Encoder blocks
    Shape6 s4in{ 2, 3, 13, 13, 13, 13 };
    run_block(st, corr4, s4in, A, Bb, S4, d_in, BASE_E4, eouts_small, ks4, st4, SUM);
    Shape6 s3in{ 2, 6, 25, 25, 25, 25 };
    run_block(st, corr3, s3in, A, Bb, S3, d_in, BASE_E3, eouts_small, ks3, st3, SUM);
    Shape6 s2in{ 2, 4, 50, 50, 50, 50 };
    run_block(st, corr2, s2in, A, Bb, S2, d_in, BASE_E2, eouts_small, ks2, st2, SUM);

    // s4: 13x13 query -> 25x25, + s3, into A
    launch_interp(st, S4, S3, A, 2, 64, 13, 13, 2, 2, 25, 25);
    // enc43: A -> Bb -> A -> Bb  (m43 in Bb)
    Shape6 f43in{ 2, 64, 25, 25, 2, 2 };
    run_block(st, A, f43in, Bb, A, Bb, d_in, BASE_E43, eouts_64, ks1, st1, SUM);
    // m43: 25x25 -> 50x50, + s2, into A
    launch_interp(st, Bb, S2, A, 2, 64, 25, 25, 2, 2, 50, 50);
    // enc32: A -> Bb -> A -> Bb  (m432 in Bb)
    Shape6 f32in{ 2, 64, 50, 50, 2, 2 };
    run_block(st, A, f32in, Bb, A, Bb, d_in, BASE_E32, eouts_64, ks1, st1, SUM);

    // mean over support + channel-duplicate concat: Bb -> A  [2,128,50,50]
    {
        long long total = 2LL * 128 * 2500;
        mean_concat_kernel<<<(unsigned)((total + 255) / 256), 256, 0, st>>>(
            Bb, A, 2500, 4, total);
    }

    // Decoder (pytree per conv: b, w)
    const float* db0 = (const float*)d_in[BASE_DEC + 0];
    const float* dw0 = (const float*)d_in[BASE_DEC + 1];
    const float* db1 = (const float*)d_in[BASE_DEC + 2];
    const float* dw1 = (const float*)d_in[BASE_DEC + 3];
    const float* db2 = (const float*)d_in[BASE_DEC + 4];
    const float* dw2 = (const float*)d_in[BASE_DEC + 5];
    const float* db3 = (const float*)d_in[BASE_DEC + 6];
    const float* dw3 = (const float*)d_in[BASE_DEC + 7];

    conv2d_dec(st, A, 2, 128, 50, 50, Bb, 64, dw0, db0, /*relu=*/1);
    conv2d_dec(st, Bb, 2, 64, 50, 50, A, 32, dw1, db1, /*relu=*/1);
    launch_interp(st, A, nullptr, Bb, 2, 32, 50, 50, 1, 1, 100, 100);  // 50 -> 100
    conv2d_dec(st, Bb, 2, 32, 100, 100, A, 16, dw2, db2, /*relu=*/1);
    conv2d_dec(st, A, 2, 16, 100, 100, (float*)d_out, 2, dw3, db3, /*relu=*/0);
}
